// SE3UPointnet_53790170415482
// MI455X (gfx1250) — matrix operand order repaired and checked by reading, not compiled, not run on hardware
//
#include <hip/hip_runtime.h>
#include <hip/hip_bf16.h>
#include <math.h>

#define EPSF 1e-8f

typedef _Float16 v16h __attribute__((ext_vector_type(16)));
typedef float    v8f  __attribute__((ext_vector_type(8)));

// ---------------------------------------------------------------------------
// order-preserving float<->uint encoding for atomicMax on floats
// ---------------------------------------------------------------------------
__device__ __forceinline__ unsigned fenc(float f) {
    int i = __float_as_int(f);
    return (i >= 0) ? ((unsigned)i + 0x80000000u) : (unsigned)(~i);
}
__device__ __forceinline__ float fdec(unsigned u) {
    int i = (u >= 0x80000000u) ? (int)(u - 0x80000000u) : ~(int)u;
    return __int_as_float(i);
}

// ---------------------------------------------------------------------------
// init per-node softmax state: m = -inf (encoded 0), z = 0
// ---------------------------------------------------------------------------
__global__ void init_mz_kernel(unsigned* __restrict__ menc, float* __restrict__ z, int N) {
    int t = blockIdx.x * blockDim.x + threadIdx.x;
    if (t < N) { menc[t] = 0u; z[t] = 0.0f; }
}

// ---------------------------------------------------------------------------
// Node transform via WMMA: q = h @ Wq, out(skip) = h @ Ws
// H2 view: row r = n*3 + i, col c;  H2[r,c] = h[n*CI*3 + c*3 + i]
// One wave = one 16-row x 16-col tile. f16 hi/lo split for ~fp32 accuracy:
//   A slots [0..15]=hi(x), [16..31]=lo(x);  B1=[Whi|Whi], B2=[Wlo|0]
//   acc += A*B1 (= (hi+lo)·Whi) ; acc += A*B2 (= hi·Wlo)  ->  ~ x·W
// ---------------------------------------------------------------------------
template<int CI, int CO>
__global__ void node_qskip_wmma(const float* __restrict__ h,
                                const float* __restrict__ Wq,
                                const float* __restrict__ Ws,
                                float* __restrict__ q,
                                float* __restrict__ outb,
                                int nNodes) {
    static_assert(CI <= 16 && CO <= 16, "channel limits");
    const int nRows = nNodes * 3;
    const int wid  = (blockIdx.x * blockDim.x + threadIdx.x) >> 5;
    const int base = wid * 16;
    if (base >= nRows) return;                 // wave-uniform: EXEC stays all-ones
    const int lane = threadIdx.x & 31;
    const int half = lane >> 4;
    const int l15  = lane & 15;

    // ---- A fragment (16-bit A 16x32 layout, ISA 7.12.2) ----
    // lane half 0: elem e<8 -> K=e,  e>=8 -> K=e+8
    // lane half 1: elem e<8 -> K=e+8, e>=8 -> K=e+16
    // => this lane covers channels [8*half, 8*half+8) as hi (slots K<16)
    //    and the same channels as lo (slots K>=16)
    const int row = base + l15;
    int n = 0, i = 0;
    if (row < nRows) { n = row / 3; i = row - n * 3; }
    v16h a;
    #pragma unroll
    for (int j = 0; j < 8; ++j) {
        int c = 8 * half + j;
        float xv = 0.0f;
        if (row < nRows && c < CI) xv = h[(size_t)n * (CI * 3) + c * 3 + i];
        _Float16 hi = (_Float16)xv;
        _Float16 lo = (_Float16)(xv - (float)hi);
        a[j]     = hi;   // slot K = c
        a[j + 8] = lo;   // slot K = c + 16
    }

    // ---- B fragments (16-bit B 32x16 layout: col per lane, K striped) ----
    const int col = l15;
    v16h bqh, bql, bsh, bsl;
    #pragma unroll
    for (int e = 0; e < 16; ++e) {
        int s = (e < 8) ? 8 * half + e : 16 + 8 * half + (e - 8); // K slot
        int c = (s < 16) ? s : s - 16;
        float wq = 0.0f, ws = 0.0f;
        if (c < CI && col < CO) { wq = Wq[c * CO + col]; ws = Ws[c * CO + col]; }
        _Float16 wqh = (_Float16)wq; _Float16 wqlo = (_Float16)(wq - (float)wqh);
        _Float16 wsh = (_Float16)ws; _Float16 wslo = (_Float16)(ws - (float)wsh);
        bqh[e] = wqh;                          // Whi in both halves
        bsh[e] = wsh;
        bql[e] = (s < 16) ? wqlo : (_Float16)0; // Wlo only vs hi(x) half
        bsl[e] = (s < 16) ? wslo : (_Float16)0;
    }

    v8f accQ = {};
    v8f accS = {};
    accQ = __builtin_amdgcn_wmma_f32_16x16x32_f16(false, a, false, bqh, (short)0, accQ, false, false);
    accQ = __builtin_amdgcn_wmma_f32_16x16x32_f16(false, a, false, bql, (short)0, accQ, false, false);
    accS = __builtin_amdgcn_wmma_f32_16x16x32_f16(false, a, false, bsh, (short)0, accS, false, false);
    accS = __builtin_amdgcn_wmma_f32_16x16x32_f16(false, a, false, bsl, (short)0, accS, false, false);

    // ---- store (f32 C/D 16x16 layout: VGPR j -> M=j (lo half) / M=j+8 (hi half)) ----
    #pragma unroll
    for (int j = 0; j < 8; ++j) {
        int r = base + j + 8 * half;
        if (r < nRows && col < CO) {
            int rn = r / 3, ri = r - rn * 3;
            size_t o = (size_t)rn * (CO * 3) + col * 3 + ri;
            q[o]    = accQ[j];
            outb[o] = accS[j];
        }
    }
}

// ---------------------------------------------------------------------------
// Edge pass 1: logits + per-destination atomic max
// ---------------------------------------------------------------------------
template<int CI, int CO>
__global__ void edge_logits_kernel(const float* __restrict__ h,
                                   const float* __restrict__ x,
                                   const int* __restrict__ src,
                                   const int* __restrict__ dst,
                                   const float* __restrict__ q,
                                   const float* __restrict__ Wk,
                                   const float* __restrict__ wrk,
                                   float* __restrict__ elog,
                                   unsigned* __restrict__ menc,
                                   int E) {
    int e = blockIdx.x * blockDim.x + threadIdx.x;
    if (e >= E) return;
    __builtin_prefetch(src + e + 2048, 0, 1);   // global_prefetch_b8 (L2-resident stream)
    __builtin_prefetch(dst + e + 2048, 0, 1);
    const int s = src[e], d = dst[e];
    float rel[3];
    #pragma unroll
    for (int i = 0; i < 3; ++i) rel[i] = x[d * 3 + i] - x[s * 3 + i];
    float hs[CI][3];
    #pragma unroll
    for (int c = 0; c < CI; ++c)
        #pragma unroll
        for (int i = 0; i < 3; ++i)
            hs[c][i] = h[(size_t)s * (CI * 3) + c * 3 + i];
    const float* qd = q + (size_t)d * (CO * 3);
    float logit = 0.0f;
    #pragma unroll
    for (int o = 0; o < CO; ++o) {
        float wr = wrk[o];
        #pragma unroll
        for (int i = 0; i < 3; ++i) {
            float kv = wr * rel[i];
            #pragma unroll
            for (int c = 0; c < CI; ++c) kv = fmaf(hs[c][i], Wk[c * CO + o], kv);
            logit = fmaf(qd[o * 3 + i], kv, logit);
        }
    }
    logit *= rsqrtf(3.0f * (float)CO);
    elog[e] = logit;
    atomicMax(&menc[d], fenc(logit));
}

// ---------------------------------------------------------------------------
// Edge pass 2: pexp = exp(logit - m[dst]); z[dst] += pexp
// ---------------------------------------------------------------------------
__global__ void edge_exp_kernel(const int* __restrict__ dst,
                                const unsigned* __restrict__ menc,
                                float* __restrict__ elog,
                                float* __restrict__ z, int E) {
    int e = blockIdx.x * blockDim.x + threadIdx.x;
    if (e >= E) return;
    int d = dst[e];
    float p = expf(elog[e] - fdec(menc[d]));
    elog[e] = p;
    atomicAdd(&z[d], p);
}

// ---------------------------------------------------------------------------
// Edge pass 3: alpha = pexp/(z+eps); out[dst] += alpha * v(e)
// ---------------------------------------------------------------------------
template<int CI, int CO>
__global__ void edge_agg_kernel(const float* __restrict__ h,
                                const float* __restrict__ x,
                                const int* __restrict__ src,
                                const int* __restrict__ dst,
                                const float* __restrict__ elog,
                                const float* __restrict__ z,
                                const float* __restrict__ Wv,
                                const float* __restrict__ wrv,
                                float* __restrict__ outb,
                                int E) {
    int e = blockIdx.x * blockDim.x + threadIdx.x;
    if (e >= E) return;
    const int s = src[e], d = dst[e];
    const float alpha = elog[e] / (z[d] + EPSF);
    float rel[3];
    #pragma unroll
    for (int i = 0; i < 3; ++i) rel[i] = x[d * 3 + i] - x[s * 3 + i];
    float hs[CI][3];
    #pragma unroll
    for (int c = 0; c < CI; ++c)
        #pragma unroll
        for (int i = 0; i < 3; ++i)
            hs[c][i] = h[(size_t)s * (CI * 3) + c * 3 + i];
    float* od = outb + (size_t)d * (CO * 3);
    #pragma unroll
    for (int o = 0; o < CO; ++o) {
        float wr = wrv[o];
        #pragma unroll
        for (int i = 0; i < 3; ++i) {
            float v = wr * rel[i];
            #pragma unroll
            for (int c = 0; c < CI; ++c) v = fmaf(hs[c][i], Wv[c * CO + o], v);
            atomicAdd(&od[o * 3 + i], alpha * v);
        }
    }
}

// ---------------------------------------------------------------------------
// GNormBias: out *= relu(|out|*a + b) / (|out| + eps)   (one thread per (n,o))
// ---------------------------------------------------------------------------
template<int CO>
__global__ void node_norm_kernel(float* __restrict__ outb,
                                 const float* __restrict__ av,
                                 const float* __restrict__ bv, int N) {
    int t = blockIdx.x * blockDim.x + threadIdx.x;
    if (t >= N * CO) return;
    int n = t / CO, o = t - n * CO;
    float* p = outb + (size_t)n * (CO * 3) + o * 3;
    float x0 = p[0], x1 = p[1], x2 = p[2];
    float nr = sqrtf(x0 * x0 + x1 * x1 + x2 * x2);
    float sc = fmaxf(nr * av[o] + bv[o], 0.0f) / (nr + EPSF);
    p[0] = x0 * sc; p[1] = x1 * sc; p[2] = x2 * sc;
}

// ---------------------------------------------------------------------------
// max pooling over KG=16 group members (elementwise over (c,i))
// ---------------------------------------------------------------------------
template<int C>
__global__ void pool_kernel(const float* __restrict__ hin,
                            const int* __restrict__ gidx,
                            float* __restrict__ hout, int N) {
    int t = blockIdx.x * blockDim.x + threadIdx.x;
    if (t >= N * C * 3) return;
    int n = t / (C * 3);
    int r = t - n * (C * 3);
    float m = -3.4e38f;
    #pragma unroll
    for (int g = 0; g < 16; ++g)
        m = fmaxf(m, hin[(size_t)gidx[n * 16 + g] * (C * 3) + r]);
    hout[t] = m;
}

// ---------------------------------------------------------------------------
// coordinate subsampling: x_next[n] = x_prev[pool_idx[n]]
// ---------------------------------------------------------------------------
__global__ void gather_x_kernel(const float* __restrict__ xin,
                                const int* __restrict__ idx,
                                float* __restrict__ xout, int N) {
    int t = blockIdx.x * blockDim.x + threadIdx.x;
    if (t >= N * 3) return;
    int n = t / 3, i = t - n * 3;
    xout[t] = xin[(size_t)idx[n] * 3 + i];
}

// ---------------------------------------------------------------------------
// 3-NN inverse-distance interpolation, C = 16 (one thread per (n,c))
// ---------------------------------------------------------------------------
__global__ void interp_kernel(const float* __restrict__ hc,
                              const float* __restrict__ xc,
                              const float* __restrict__ xf,
                              const int* __restrict__ fp,
                              float* __restrict__ hf, int Nf) {
    int t = blockIdx.x * blockDim.x + threadIdx.x;
    if (t >= Nf * 16) return;
    int n = t >> 4, c = t & 15;
    int i0 = fp[n * 3 + 0], i1 = fp[n * 3 + 1], i2 = fp[n * 3 + 2];
    float fx = xf[n * 3 + 0], fy = xf[n * 3 + 1], fz = xf[n * 3 + 2];
    float dx0 = fx - xc[i0 * 3], dy0 = fy - xc[i0 * 3 + 1], dz0 = fz - xc[i0 * 3 + 2];
    float dx1 = fx - xc[i1 * 3], dy1 = fy - xc[i1 * 3 + 1], dz1 = fz - xc[i1 * 3 + 2];
    float dx2 = fx - xc[i2 * 3], dy2 = fy - xc[i2 * 3 + 1], dz2 = fz - xc[i2 * 3 + 2];
    float w0 = 1.0f / (dx0 * dx0 + dy0 * dy0 + dz0 * dz0 + EPSF);
    float w1 = 1.0f / (dx1 * dx1 + dy1 * dy1 + dz1 * dz1 + EPSF);
    float w2 = 1.0f / (dx2 * dx2 + dy2 * dy2 + dz2 * dz2 + EPSF);
    float inv = 1.0f / (w0 + w1 + w2);
    w0 *= inv; w1 *= inv; w2 *= inv;
    #pragma unroll
    for (int i = 0; i < 3; ++i) {
        hf[(size_t)n * 48 + c * 3 + i] =
            w0 * hc[(size_t)i0 * 48 + c * 3 + i] +
            w1 * hc[(size_t)i1 * 48 + c * 3 + i] +
            w2 * hc[(size_t)i2 * 48 + c * 3 + i];
    }
}

// ---------------------------------------------------------------------------
// host-side conv_layer orchestration
// ---------------------------------------------------------------------------
template<int CI, int CO>
static void run_conv_layer(const float* h, const float* x,
                           const int* src, const int* dst,
                           const float* const* L,   // Wq,Wk,Wv,Ws,wrk,wrv,a,b
                           float* q, float* outb, float* elog,
                           unsigned* menc, float* z,
                           int N, int E, hipStream_t stream) {
    const int T = 256;
    init_mz_kernel<<<(N + T - 1) / T, T, 0, stream>>>(menc, z, N);
    int tiles  = (N * 3 + 15) / 16;
    int blocks = (tiles + 7) / 8;                 // 8 waves / 256-thread block
    node_qskip_wmma<CI, CO><<<blocks, T, 0, stream>>>(h, L[0], L[3], q, outb, N);
    edge_logits_kernel<CI, CO><<<(E + T - 1) / T, T, 0, stream>>>(h, x, src, dst, q, L[1], L[4], elog, menc, E);
    edge_exp_kernel<<<(E + T - 1) / T, T, 0, stream>>>(dst, menc, elog, z, E);
    edge_agg_kernel<CI, CO><<<(E + T - 1) / T, T, 0, stream>>>(h, x, src, dst, elog, z, L[2], L[5], outb, E);
    node_norm_kernel<CO><<<(N * CO + T - 1) / T, T, 0, stream>>>(outb, L[6], L[7], N);
}

extern "C" void kernel_launch(void* const* d_in, const int* in_sizes, int n_in,
                              void* d_out, int out_size, void* d_ws, size_t ws_size,
                              hipStream_t stream) {
    (void)in_sizes; (void)out_size; (void)ws_size;
    const int N0 = 200000, N1 = 40000, N2 = 8000, N3 = 1600, N4 = 320;
    const int E0 = N0 * 32, E1 = N1 * 32, E2 = N2 * 32, E3 = N3 * 32;

    const float* x0 = (const float*)d_in[0];
    const float* v0 = (const float*)d_in[1];
    const int* es[4] = { (const int*)d_in[2], (const int*)d_in[4], (const int*)d_in[6], (const int*)d_in[8] };
    const int* ed[4] = { (const int*)d_in[3], (const int*)d_in[5], (const int*)d_in[7], (const int*)d_in[9] };
    const int* pool1 = (const int*)d_in[10]; const int* g1 = (const int*)d_in[11]; const int* fp1 = (const int*)d_in[12];
    const int* pool2 = (const int*)d_in[13]; const int* g2 = (const int*)d_in[14]; const int* fp2 = (const int*)d_in[15];
    const int* pool3 = (const int*)d_in[16]; const int* g3 = (const int*)d_in[17]; const int* fp3 = (const int*)d_in[18];
    const int* pool4 = (const int*)d_in[19]; const int* g4 = (const int*)d_in[20]; const int* fp4 = (const int*)d_in[21];

    // ---- resolve the 64 parameter leaves (separate leaves OR one blob) ----
    static const int cin_[4]  = { 1, 2, 4, 8 };
    static const int cout_[4] = { 2, 4, 8, 16 };
    const float* leaf[64];
    if (n_in >= 22 + 64) {
        for (int j = 0; j < 64; ++j) leaf[j] = (const float*)d_in[22 + j];
    } else {
        const float* blob = (const float*)d_in[22];
        size_t off = 0; int j = 0;
        for (int l = 0; l < 4; ++l)
            for (int t = 0; t < 2; ++t) {
                int ci = t ? cout_[l] : cin_[l]; int co = cout_[l];
                for (int m = 0; m < 4; ++m) { leaf[j++] = blob + off; off += (size_t)ci * co; }
                for (int m = 0; m < 4; ++m) { leaf[j++] = blob + off; off += co; }
            }
    }
    const float* const* P0 = &leaf[0];   const float* const* P1 = &leaf[8];
    const float* const* P2 = &leaf[16];  const float* const* P3 = &leaf[24];
    const float* const* P4 = &leaf[32];  const float* const* P5 = &leaf[40];
    const float* const* P6 = &leaf[48];  const float* const* P7 = &leaf[56];

    // ---- workspace layout (floats), ~49 MB total ----
    float* p = (float*)d_ws;
    float* hA = p;   p += 2000000;
    float* hB = p;   p += 2000000;
    float* q  = p;   p += 1250000;
    float* el = p;   p += 6400000;
    float* z  = p;   p += 200000;
    unsigned* menc = (unsigned*)p; p += 200000;
    float* x1 = p; p += (size_t)N1 * 3;
    float* x2 = p; p += (size_t)N2 * 3;
    float* x3 = p; p += (size_t)N3 * 3;
    float* x4 = p; p += (size_t)N4 * 3;

    const int T = 256;

    // ---- Level 0 ----
    run_conv_layer<1, 2>(v0, x0, es[0], ed[0], P0, q, hA, el, menc, z, N0, E0, stream);
    run_conv_layer<2, 2>(hA, x0, es[0], ed[0], P1, q, hB, el, menc, z, N0, E0, stream);
    gather_x_kernel<<<(N1 * 3 + T - 1) / T, T, 0, stream>>>(x0, pool1, x1, N1);
    pool_kernel<2><<<(N1 * 2 * 3 + T - 1) / T, T, 0, stream>>>(hB, g1, hA, N1);

    // ---- Level 1 ----
    run_conv_layer<2, 4>(hA, x1, es[1], ed[1], P2, q, hB, el, menc, z, N1, E1, stream);
    run_conv_layer<4, 4>(hB, x1, es[1], ed[1], P3, q, hA, el, menc, z, N1, E1, stream);
    gather_x_kernel<<<(N2 * 3 + T - 1) / T, T, 0, stream>>>(x1, pool2, x2, N2);
    pool_kernel<4><<<(N2 * 4 * 3 + T - 1) / T, T, 0, stream>>>(hA, g2, hB, N2);

    // ---- Level 2 ----
    run_conv_layer<4, 8>(hB, x2, es[2], ed[2], P4, q, hA, el, menc, z, N2, E2, stream);
    run_conv_layer<8, 8>(hA, x2, es[2], ed[2], P5, q, hB, el, menc, z, N2, E2, stream);
    gather_x_kernel<<<(N3 * 3 + T - 1) / T, T, 0, stream>>>(x2, pool3, x3, N3);
    pool_kernel<8><<<(N3 * 8 * 3 + T - 1) / T, T, 0, stream>>>(hB, g3, hA, N3);

    // ---- Level 3 ----
    run_conv_layer<8, 16>(hA, x3, es[3], ed[3], P6, q, hB, el, menc, z, N3, E3, stream);
    run_conv_layer<16, 16>(hB, x3, es[3], ed[3], P7, q, hA, el, menc, z, N3, E3, stream);
    gather_x_kernel<<<(N4 * 3 + T - 1) / T, T, 0, stream>>>(x3, pool4, x4, N4);
    pool_kernel<16><<<(N4 * 16 * 3 + T - 1) / T, T, 0, stream>>>(hA, g4, hB, N4);

    // ---- Feature-propagation (interp) chain ----
    interp_kernel<<<(N3 * 16 + T - 1) / T, T, 0, stream>>>(hB, x4, x3, fp4, hA, N3);
    interp_kernel<<<(N2 * 16 + T - 1) / T, T, 0, stream>>>(hA, x3, x2, fp3, hB, N2);
    interp_kernel<<<(N1 * 16 + T - 1) / T, T, 0, stream>>>(hB, x2, x1, fp2, hA, N1);
    interp_kernel<<<(N0 * 16 + T - 1) / T, T, 0, stream>>>(hA, x1, x0, fp1, (float*)d_out, N0);
}
